// WindowMSA_79491254714886
// MI455X (gfx1250) — matrix operand order repaired and checked by reading, not compiled, not run on hardware
//
#include <hip/hip_runtime.h>
#include <hip/hip_bf16.h>

// ---------------- problem constants (from reference) ----------------
#define WIN    8
#define HEADS  4
#define CCH    96          // channels
#define HD     24          // head dim
#define NTOK   64          // tokens per window
#define HIMG   256
#define WIMG   256
#define EPSV   1e-5f
#define NWIN   8192        // 8 images * 32 * 32 windows

typedef __attribute__((ext_vector_type(16))) _Float16 v16h;
typedef __attribute__((ext_vector_type(8)))  _Float16 v8h;
typedef __attribute__((ext_vector_type(8)))  float    v8f;

// ---------------- WMMA helpers (CDNA5 ISA 7.12.2 layouts) ----------------
__device__ inline v8f wmma_f16(v16h a, v16h b, v8f c) {
  // D = A(16x32 f16) * B(32x16 f16) + C(16x16 f32)
  return __builtin_amdgcn_wmma_f32_16x16x32_f16(
      /*neg_a=*/false, a, /*neg_b=*/false, b,
      /*c_mod=*/(short)0, c, /*reuse_a=*/false, /*reuse_b=*/false);
}

// A-fragment from f16 matrix, row-major, leading dim ld (halfs).
// lane m=lane&15, kg=lane>>4; elems 0..15 -> k=kg*16+i (this code's k order, same for B)
__device__ inline v16h frag_a_lds(const _Float16* base, int ld, int lane) {
  int m = lane & 15, kg = lane >> 4;
  const _Float16* p = base + m * ld;
  v8h c0 = *(const v8h*)(p + kg * 16);
  v8h c1 = *(const v8h*)(p + kg * 16 + 8);
  v16h a;
#pragma unroll
  for (int i = 0; i < 8; ++i) { a[i] = c0[i]; a[8 + i] = c1[i]; }
  return a;
}

// A-fragment for Q (valid k < 24, pad to 32): only lanes 16..31 second chunk
// (k=24..31) is invalid -> vector loads + lane-uniform select.
__device__ inline v16h frag_a_q(const _Float16* base, int lane) {
  int m = lane & 15, kg = lane >> 4;
  const _Float16* p = base + m * 192;
  v8h c0 = *(const v8h*)(p + kg * 16);         // k=kg*16..+7, always < 24
  v8h c1 = *(const v8h*)(p + kg * 16 + 8);     // kg=0: k=8..15 valid; kg=1: 24..31 pad
  v16h a;
#pragma unroll
  for (int i = 0; i < 8; ++i) {
    a[i]     = c0[i];
    a[8 + i] = kg ? (_Float16)0.0f : c1[i];
  }
  return a;
}

// B-fragment for K (32x16, valid k < 24): lane n=lane&15, kh=lane>>4; k=kh*16+i.
// kh=0: all 16 valid; kh=1: first 8 (k=16..23) valid, last 8 pad.
__device__ inline v16h frag_b_k(const _Float16* base, int lane) {
  int n = lane & 15, kh = lane >> 4;
  const _Float16* p = base + n * 192 + kh * 16;
  v8h c0 = *(const v8h*)(p);
  v8h c1 = *(const v8h*)(p + 8);
  v16h b;
#pragma unroll
  for (int i = 0; i < 8; ++i) {
    b[i]     = c0[i];
    b[8 + i] = kh ? (_Float16)0.0f : c1[i];
  }
  return b;
}

// B-fragment (32x16, K x N) from row-major [N][K] storage (weights [d][c]):
// each lane reads 32 contiguous bytes.
__device__ inline v16h frag_b_rowmajor(const _Float16* base, int ld, int lane) {
  int n = lane & 15, kh = lane >> 4;
  return *(const v16h*)(base + n * ld + kh * 16);
}

// ---------------- weight f32 -> f16 prologue ----------------
__global__ __launch_bounds__(256) void convert_weights(
    const float* __restrict__ wqkv,   // [288][96]
    const float* __restrict__ wout,   // [96][96]
    const float* __restrict__ wproj,  // [96][96]
    _Float16* __restrict__ dst)
{
  int i = blockIdx.x * 256 + threadIdx.x;            // 108 blocks -> 27648
  if (i < 288 * 96) dst[i] = (_Float16)wqkv[i];
  if (i < 96 * 96) {
    dst[288 * 96 + i]           = (_Float16)wout[i];
    dst[288 * 96 + 96 * 96 + i] = (_Float16)wproj[i];
  }
}

// ---------------- fused window-MSA kernel: 1 block = 1 window ----------------
__global__ __launch_bounds__(256) void winmsa_kernel(
    const float* __restrict__ x,       // [8,96,256,256]
    const float* __restrict__ ln_w, const float* __restrict__ ln_b,
    const float* __restrict__ in_b,    // [288]
    const float* __restrict__ out_b,   // [96]
    const float* __restrict__ proj_b,  // [96]
    const _Float16* __restrict__ wqkv, // [288][96] f16
    const _Float16* __restrict__ wout, // [96][96]  f16
    const _Float16* __restrict__ wproj,// [96][96]  f16
    float* __restrict__ out)
{
  // 64 KB LDS, phase-aliased:
  //  A [0,12288)      : tln f16 [64][96]  -> later o (attn*V output) f16 [64][96]
  //  B [12288,36864)  : traw f32 [96][64] -> qk f16 [64][192] -> outb f32 [96][64]
  //  C [36864,49152)  : vT f16 [96][64]
  //  D [49152,65536)  : LN scratch -> per-head scores f32 [64][64] (softmaxed
  //                     in-place to attn f16 pitch 128) -> o2 f16 [64][96]
  __shared__ __align__(32) char smem[65536];
  _Float16* tln  = (_Float16*)(smem);
  _Float16* oacc = (_Float16*)(smem);
  float*    traw = (float*)(smem + 12288);      // [c][64] layout
  _Float16* qk   = (_Float16*)(smem + 12288);   // [n][192]  (Q | K)
  float*    outb = (float*)(smem + 12288);      // [c][64] layout
  _Float16* vT   = (_Float16*)(smem + 36864);   // [d][64]
  float*    sc   = (float*)(smem + 49152);      // [64][64] per head
  _Float16* attn = (_Float16*)(smem + 49152);   // pitch 128 halfs over sc rows
  _Float16* o2   = (_Float16*)(smem + 49152);   // [64][96]
  float*    lred = (float*)(smem + 49152);      // [64][4][2] partial sums
  float*    musig= lred + 512;                  // [64][2] mean, rstd

  const int tid  = threadIdx.x;
  const int lane = tid & 31;
  const int wave = tid >> 5;

  const int wid = blockIdx.x;                 // window index = b*1024 + wy*32 + wx
  const int b   = wid >> 10;
  const int wy  = (wid >> 5) & 31;
  const int wx  = wid & 31;
  const int HW  = HIMG * WIMG;
  const int gbase = (b * CCH) * HW + (wy * WIN) * WIMG + wx * WIN;

  // ---- phase 1: gather window (float4, [c][n] staging), LayerNorm over c ----
#pragma unroll
  for (int it = 0; it < 6; ++it) {
    int e4 = it * 256 + tid;                  // 1536 float4s
    int c  = e4 >> 4;
    int n0 = (e4 & 15) * 4;                   // 4 consecutive tokens in one image row
    int y  = n0 >> 3, xx = n0 & 7;
    *(float4*)(traw + c * 64 + n0) =
        *(const float4*)(x + gbase + c * HW + y * WIMG + xx);
  }
  __syncthreads();
  {
    int n = tid >> 2, p = tid & 3;            // 4 partial reducers per token
    float s = 0.f, ss = 0.f;
#pragma unroll
    for (int i = 0; i < 24; ++i) {
      float v = traw[(p * 24 + i) * 64 + n];
      s += v; ss += v * v;
    }
    lred[(n * 4 + p) * 2 + 0] = s;
    lred[(n * 4 + p) * 2 + 1] = ss;
  }
  __syncthreads();
  if (tid < 64) {
    float s = 0.f, ss = 0.f;
#pragma unroll
    for (int p = 0; p < 4; ++p) { s += lred[(tid * 4 + p) * 2]; ss += lred[(tid * 4 + p) * 2 + 1]; }
    float mu  = s * (1.0f / 96.0f);
    float var = ss * (1.0f / 96.0f) - mu * mu;
    musig[tid * 2 + 0] = mu;
    musig[tid * 2 + 1] = rsqrtf(var + EPSV);
  }
  __syncthreads();
#pragma unroll
  for (int it = 0; it < 24; ++it) {
    int e = it * 256 + tid;
    int c = e >> 6, n = e & 63;
    float v = (traw[c * 64 + n] - musig[n * 2]) * musig[n * 2 + 1] * ln_w[c] + ln_b[c];
    tln[n * CCH + c] = (_Float16)v;
  }
  __syncthreads();

  // ---- phase 2: QKV GEMM  [64x96] x [96x288] ; V written transposed ----
  for (int j = wave; j < 72; j += 8) {        // 4 Mtiles x 18 Ntiles
    int mt = j / 18, nt = j % 18;
    v8f acc = {};
#pragma unroll
    for (int kk = 0; kk < 3; ++kk) {
      v16h a  = frag_a_lds(tln + mt * 16 * CCH + kk * 32, CCH, lane);
      v16h bf = frag_b_rowmajor(wqkv + nt * 16 * CCH + kk * 32, CCH, lane);
      acc = wmma_f16(a, bf, acc);
    }
    int n  = nt * 16 + (lane & 15);
    int mg = lane >> 4;
    float bias = in_b[n];
    if (n < 192) {                            // Q | K -> qk[token][0..191]
#pragma unroll
      for (int r = 0; r < 8; ++r)
        qk[(mt * 16 + mg * 8 + r) * 192 + n] = (_Float16)(acc[r] + bias);
    } else {                                  // V -> vT[d][token] (contiguous b128)
#pragma unroll
      for (int r = 0; r < 8; ++r)
        vT[(n - 192) * 64 + (mt * 16 + mg * 8 + r)] = (_Float16)(acc[r] + bias);
    }
  }
  __syncthreads();

  // ---- phases 3-5, per head: scores, softmax, attn*V ----
  const float scale = 0.2041241452319315f;    // 1/sqrt(24)
  for (int h = 0; h < HEADS; ++h) {
    // scores = Q K^T * scale  (K-dim 24 zero-padded to 32)
    for (int j = wave; j < 16; j += 8) {      // 4x4 16x16 tiles
      int mt = j >> 2, nt = j & 3;
      v16h a  = frag_a_q(qk + mt * 16 * 192 + h * HD, lane);
      v16h bf = frag_b_k(qk + nt * 16 * 192 + 96 + h * HD, lane);
      v8f acc = {};
      acc = wmma_f16(a, bf, acc);
      int n = lane & 15, mg = lane >> 4;
#pragma unroll
      for (int r = 0; r < 8; ++r)
        sc[(mt * 16 + mg * 8 + r) * 64 + nt * 16 + n] = acc[r] * scale;
    }
    __syncthreads();
    // softmax per query row; attn f16 written in-place over sc (pitch 128 halfs)
    if (tid < 64) {
      int q = tid;
      float mx = -1e30f;
      for (int m = 0; m < 64; ++m) mx = fmaxf(mx, sc[q * 64 + m]);
      float sum = 0.f;
      for (int m = 0; m < 64; ++m) { float e = __expf(sc[q * 64 + m] - mx); sc[q * 64 + m] = e; sum += e; }
      float inv = 1.0f / sum;
      for (int m = 0; m < 64; ++m)            // write byte 2m while reading byte 4m: safe ascending
        attn[q * 128 + m] = (_Float16)(sc[q * 64 + m] * inv);
    }
    __syncthreads();
    // o[:, h*24 : h*24+24] = attn @ V   (K=64, N padded 24->32)
    {
      int mt = wave >> 1, nt = wave & 1;      // 8 waves = 8 tile jobs
      v8f acc = {};
#pragma unroll
      for (int ks = 0; ks < 2; ++ks) {
        v16h a = frag_a_lds(attn + mt * 16 * 128 + ks * 32, 128, lane);
        int dg = h * HD + nt * 16 + (lane & 15);
        if (dg > 95) dg = 95;                 // clamp pad columns (results discarded)
        v16h bf = *(const v16h*)(vT + dg * 64 + ks * 32 + (lane >> 4) * 16);
        acc = wmma_f16(a, bf, acc);
      }
      int d = nt * 16 + (lane & 15), mg = lane >> 4;
      if (d < HD) {
#pragma unroll
        for (int r = 0; r < 8; ++r)
          oacc[(mt * 16 + mg * 8 + r) * CCH + h * HD + d] = (_Float16)acc[r];
      }
    }
    __syncthreads();
  }

  // ---- phase 6: out_proj  [64x96] x [96x96] ----
  for (int j = wave; j < 24; j += 8) {        // 4x6 tiles
    int mt = j / 6, nt = j % 6;
    v8f acc = {};
#pragma unroll
    for (int kk = 0; kk < 3; ++kk) {
      v16h a  = frag_a_lds(oacc + mt * 16 * CCH + kk * 32, CCH, lane);
      v16h bf = frag_b_rowmajor(wout + nt * 16 * CCH + kk * 32, CCH, lane);
      acc = wmma_f16(a, bf, acc);
    }
    int n = nt * 16 + (lane & 15), mg = lane >> 4;
    float bias = out_b[n];
#pragma unroll
    for (int r = 0; r < 8; ++r)
      o2[(mt * 16 + mg * 8 + r) * CCH + n] = (_Float16)(acc[r] + bias);
  }
  __syncthreads();

  // ---- phase 7: final proj -> f32 tile in [c][n] layout (contiguous stores) ----
  for (int j = wave; j < 24; j += 8) {
    int mt = j / 6, nt = j % 6;
    v8f acc = {};
#pragma unroll
    for (int kk = 0; kk < 3; ++kk) {
      v16h a  = frag_a_lds(o2 + mt * 16 * CCH + kk * 32, CCH, lane);
      v16h bf = frag_b_rowmajor(wproj + nt * 16 * CCH + kk * 32, CCH, lane);
      acc = wmma_f16(a, bf, acc);
    }
    int c  = nt * 16 + (lane & 15);
    int t0 = mt * 16 + (lane >> 4) * 8;
    float bias = proj_b[c];
#pragma unroll
    for (int r = 0; r < 8; ++r)               // 8 consecutive floats -> b128 stores
      outb[c * 64 + t0 + r] = acc[r] + bias;
  }
  __syncthreads();

  // ---- phase 8: scatter window back to [B,C,H,W] (float4) ----
#pragma unroll
  for (int it = 0; it < 6; ++it) {
    int e4 = it * 256 + tid;
    int c  = e4 >> 4;
    int n0 = (e4 & 15) * 4;
    int y  = n0 >> 3, xx = n0 & 7;
    *(float4*)(out + gbase + c * HW + y * WIMG + xx) =
        *(const float4*)(outb + c * 64 + n0);
  }
}

extern "C" void kernel_launch(void* const* d_in, const int* in_sizes, int n_in,
                              void* d_out, int out_size, void* d_ws, size_t ws_size,
                              hipStream_t stream) {
  const float* x    = (const float*)d_in[0];
  const float* ln_w = (const float*)d_in[1];
  const float* ln_b = (const float*)d_in[2];
  const float* ipw  = (const float*)d_in[3];
  const float* ipb  = (const float*)d_in[4];
  const float* opw  = (const float*)d_in[5];
  const float* opb  = (const float*)d_in[6];
  const float* pw   = (const float*)d_in[7];
  const float* pb   = (const float*)d_in[8];

  _Float16* w16 = (_Float16*)d_ws;            // 92160 bytes of f16 weights
  convert_weights<<<108, 256, 0, stream>>>(ipw, opw, pw, w16);

  winmsa_kernel<<<NWIN, 256, 0, stream>>>(
      x, ln_w, ln_b, ipb, opb, pb,
      w16,                       // [288][96] in_proj
      w16 + 288 * 96,            // [96][96]  out_proj
      w16 + 288 * 96 + 96 * 96,  // [96][96]  proj
      (float*)d_out);
}
